// BinarizeConv2dSDP_82910048682620
// MI455X (gfx1250) — 8-bit matrix operand order repaired and checked by reading, compiled, not run on hardware
//
#include <hip/hip_runtime.h>
#include <cstdint>

typedef __attribute__((ext_vector_type(16))) int   v16i;
typedef __attribute__((ext_vector_type(8)))  float v8f;

#define FP8_P1 0x38u
#define FP8_M1 0xB8u

#define CIN     256
#define COUT    256
#define HW      784
#define KTOT    2304
#define NPOS    50176
#define ACT_SZ  12845056
#define WQ_SZ   589824
#define AROW    144
#define NKT     (KTOT / 128)

static __device__ __forceinline__ uint8_t sgn_fp8(float v) {
    return v > 0.f ? (uint8_t)FP8_P1 : (v < 0.f ? (uint8_t)FP8_M1 : (uint8_t)0);
}

static __device__ __forceinline__ unsigned lds_off(const void* p) {
    return (unsigned)(uintptr_t)(__attribute__((address_space(3))) const void*)p;
}

__global__ __launch_bounds__(256) void binz_act_nhwc(const float* __restrict__ x,
                                                     uint8_t* __restrict__ actT) {
    int tid = blockIdx.x * 256 + threadIdx.x;
    int pos = tid >> 6;
    int c0  = (tid & 63) * 4;
    int b   = pos / HW;
    int hw  = pos - b * HW;
    const float* xs = x + b * (CIN * HW) + c0 * HW + hw;
    uchar4 s;
    s.x = sgn_fp8(xs[0]);
    s.y = sgn_fp8(xs[HW]);
    s.z = sgn_fp8(xs[2 * HW]);
    s.w = sgn_fp8(xs[3 * HW]);
    *(uchar4*)(actT + pos * 256 + c0) = s;
}

__global__ __launch_bounds__(256) void binz_w(const float* __restrict__ M,
                                              const float* __restrict__ Z,
                                              const float* __restrict__ rv,
                                              uint8_t* __restrict__ Wq,
                                              uint8_t* __restrict__ zpad) {
    int n = blockIdx.x * 256 + threadIdx.x;
    if (n < 8) {
        uint4 z = {0u, 0u, 0u, 0u};
        *(uint4*)(zpad + n * 16) = z;
    }
    float w = M[n];
#pragma unroll
    for (int k = 0; k < 8; ++k)
        w = fmaf(rv[k], Z[k * (COUT * KTOT) + n], w);
    int o   = n / KTOT;
    int rem = n - o * KTOT;
    int c   = rem / 9;
    int r9  = rem - c * 9;
    Wq[o * KTOT + r9 * 256 + c] = sgn_fp8(w);
}

__global__ __launch_bounds__(256) void bconv_wmma(const uint8_t* __restrict__ Wq,
                                                  const uint8_t* __restrict__ actT,
                                                  const float* __restrict__ alpha,
                                                  const uint8_t* __restrict__ zpad,
                                                  float* __restrict__ out) {
    __shared__ __align__(16) uint8_t Atile[2][128 * AROW];

    const int t    = threadIdx.x;
    const int lane = t & 31;
    const int wave = t >> 5;
    const int wm   = wave >> 1;
    const int wn   = wave & 1;
    const int half = lane >> 4;
    const int l16  = lane & 15;

    const int pblk = blockIdx.x * 128;
    const int oblk = blockIdx.y * 128;

    const int ldA_row  = t >> 1;
    const int ldA_koff = (t & 1) * 64;
    const uint8_t* wsrc   = Wq + (oblk + ldA_row) * KTOT + ldA_koff;
    const int      ldAoff = ldA_row * AROW + ldA_koff;
    const unsigned ldsA0  = lds_off(&Atile[0][ldAoff]);
    const unsigned ldsA1  = lds_off(&Atile[1][ldAoff]);

    const uint8_t* bptr[4];
    int hj[4], wj[4], obix[4];
#pragma unroll
    for (int j = 0; j < 4; ++j) {
        int p  = pblk + wn * 64 + j * 16 + l16;
        int b  = p / HW;
        int hw = p - b * HW;
        hj[j]  = hw / 28;
        wj[j]  = hw - 28 * hj[j];
        bptr[j] = actT + (size_t)p * 256;
        obix[j] = b * (COUT * HW) + hw;
    }

    v8f zero = {};
    v8f acc[2][4];
#pragma unroll
    for (int i = 0; i < 2; ++i)
#pragma unroll
        for (int j = 0; j < 4; ++j) acc[i][j] = zero;

#define ASYNC_A64(ldsoff, gsrc)                                                      \
    asm volatile("global_load_async_to_lds_b128 %0, %1, off"           ::            \
                 "v"(ldsoff), "v"(gsrc) : "memory");                                 \
    asm volatile("global_load_async_to_lds_b128 %0, %1, off offset:16" ::            \
                 "v"(ldsoff), "v"(gsrc) : "memory");                                 \
    asm volatile("global_load_async_to_lds_b128 %0, %1, off offset:32" ::            \
                 "v"(ldsoff), "v"(gsrc) : "memory");                                 \
    asm volatile("global_load_async_to_lds_b128 %0, %1, off offset:48" ::            \
                 "v"(ldsoff), "v"(gsrc) : "memory");

    ASYNC_A64(ldsA0, wsrc)
    asm volatile("s_wait_asynccnt 0x0" ::: "memory");
    __syncthreads();

    for (int kt = 0; kt < NKT; ++kt) {
        const int cur = kt & 1;
        const int r9  = kt >> 1;
        const int kh  = r9 / 3;
        const int kw  = r9 - 3 * kh;
        const int dh  = kh - 1;
        const int dw  = kw - 1;
        const int c0  = (kt & 1) << 7;
        const int boff = (dh * 28 + dw) * 256 + c0 + half * 16;

        if (kt + 1 < NKT) {
            const uint8_t* src = wsrc + (kt + 1) * 128;
            const unsigned dst = cur ? ldsA0 : ldsA1;
            ASYNC_A64(dst, src)
        }

        v16i afrag[2];
#pragma unroll
        for (int i = 0; i < 2; ++i) {
            const uint8_t* ab = &Atile[cur][(wm * 32 + i * 16 + l16) * AROW + half * 16];
#pragma unroll
            for (int b2 = 0; b2 < 2; ++b2)
#pragma unroll
                for (int g = 0; g < 4; ++g) {
                    int2 d = *(const int2*)(ab + b2 * 64 + (g >> 1) * 32 + (g & 1) * 8);
                    afrag[i][b2 * 8 + 2 * g]     = d.x;
                    afrag[i][b2 * 8 + 2 * g + 1] = d.y;
                }
        }

#pragma unroll
        for (int j = 0; j < 4; ++j) {
            int ih = hj[j] + dh;
            int iw = wj[j] + dw;
            bool inb = ((unsigned)ih < 28u) & ((unsigned)iw < 28u);
            const uint8_t* src = inb ? (bptr[j] + boff) : zpad;
            int4 d0 = *(const int4*)(src + 0);
            int4 d1 = *(const int4*)(src + 32);
            int4 d2 = *(const int4*)(src + 64);
            int4 d3 = *(const int4*)(src + 96);
            v16i bfrag;
            bfrag[0]  = d0.x; bfrag[1]  = d0.y; bfrag[2]  = d0.z; bfrag[3]  = d0.w;
            bfrag[4]  = d1.x; bfrag[5]  = d1.y; bfrag[6]  = d1.z; bfrag[7]  = d1.w;
            bfrag[8]  = d2.x; bfrag[9]  = d2.y; bfrag[10] = d2.z; bfrag[11] = d2.w;
            bfrag[12] = d3.x; bfrag[13] = d3.y; bfrag[14] = d3.z; bfrag[15] = d3.w;
#pragma unroll
            for (int i = 0; i < 2; ++i)
                acc[i][j] = __builtin_amdgcn_wmma_f32_16x16x128_fp8_fp8(
                    afrag[i], bfrag, (short)0, acc[i][j], false, false);
        }

        asm volatile("s_wait_asynccnt 0x0" ::: "memory");
        __syncthreads();
    }

#pragma unroll
    for (int i = 0; i < 2; ++i) {
        int obase = oblk + wm * 32 + i * 16 + 8 * half;
#pragma unroll
        for (int r = 0; r < 8; ++r) {
            int o = obase + r;
            float a = alpha[o];
            int ooff = o * HW;
#pragma unroll
            for (int j = 0; j < 4; ++j)
                out[obix[j] + ooff] = acc[i][j][r] * a;
        }
    }
#undef ASYNC_A64
}

extern "C" void kernel_launch(void* const* d_in, const int* in_sizes, int n_in,
                              void* d_out, int out_size, void* d_ws, size_t ws_size,
                              hipStream_t stream) {
    (void)in_sizes; (void)n_in; (void)out_size; (void)ws_size;
    const float* x     = (const float*)d_in[0];
    const float* M     = (const float*)d_in[1];
    const float* Z     = (const float*)d_in[2];
    const float* alpha = (const float*)d_in[3];
    const float* rv    = (const float*)d_in[4];
    float* out = (float*)d_out;

    uint8_t* actT = (uint8_t*)d_ws;
    uint8_t* Wq   = actT + ACT_SZ;
    uint8_t* zpad = Wq + WQ_SZ;

    binz_act_nhwc<<<(NPOS * 64) / 256, 256, 0, stream>>>(x, actT);
    binz_w<<<(COUT * KTOT) / 256, 256, 0, stream>>>(M, Z, rv, Wq, zpad);

    dim3 grid(NPOS / 128, COUT / 128);
    bconv_wmma<<<grid, 256, 0, stream>>>(Wq, actT, alpha, zpad, out);
}
